// ActorCriticBatch_68977174773816
// MI455X (gfx1250) — matrix operand order repaired and checked by reading, not compiled, not run on hardware
//
#include <hip/hip_runtime.h>

// ---------------------------------------------------------------------------
// GIN actor-critic forward for MI455X (gfx1250, wave32, WMMA).
// All matmuls run on v_wmma_f32_16x16x32_f16 (f16 inputs, f32 accumulate).
// ---------------------------------------------------------------------------

typedef __attribute__((ext_vector_type(16))) _Float16 v16h;
typedef __attribute__((ext_vector_type(8)))  float    v8f;

#define BG   64        // graphs
#define NN   512       // nodes per graph
#define DD   32        // feature dim
#define NT   (BG*NN)   // 32768 total nodes
#define ET   (NN*32*BG)// 1048576 directed edges total (DEG=32)
#define EG   (NN*32)   // 16384 edges per graph
#define BN_EPS 1e-5f

__device__ __forceinline__ v8f wmma_f16(v16h a, v16h b, v8f c) {
  // D = A(16x32 f16) * B(32x16 f16) + C(16x16 f32)
  return __builtin_amdgcn_wmma_f32_16x16x32_f16(
      false, a, false, b, (short)0, c, false, false);
}

// A-fragment loader: lane holds row m=lane&15; elements 0..15 = K[16*hf .. +15]
// (contiguous k order, matching the staged B fragments).
__device__ __forceinline__ v16h load_a(const _Float16* __restrict__ rowp, int hf) {
  v16h a;
#pragma unroll
  for (int j = 0; j < 8; ++j) {
    a[j]     = rowp[16 * hf + j];
    a[8 + j] = rowp[16 * hf + 8 + j];
  }
  return a;
}

// ---------------------------------------------------------------------------
__global__ void __launch_bounds__(256) k_zero(float* __restrict__ p, int n) {
  int i = blockIdx.x * 256 + threadIdx.x;
  if (i < n) p[i] = 0.0f;
}

// Symmetrized edge aggregation: agg[dst]+=f[src] and agg[src]+=f[dst].
// One thread per (edge, 8-float chunk).
__global__ void __launch_bounds__(256)
k_scatter(const float* __restrict__ f, const int* __restrict__ src,
          const int* __restrict__ dst, float* __restrict__ agg) {
  int gid = blockIdx.x * 256 + threadIdx.x;
  int e = gid >> 2, c = gid & 3;
  if (e >= ET) return;
  int s = src[e], d = dst[e];
  const float4* fs = (const float4*)(f + s * DD + c * 8);
  const float4* fd = (const float4*)(f + d * DD + c * 8);
  float4 a0 = fs[0], a1 = fs[1];
  float4 b0 = fd[0], b1 = fd[1];
  float* ad = agg + d * DD + c * 8;
  float* as = agg + s * DD + c * 8;
  atomicAdd(ad + 0, a0.x); atomicAdd(ad + 1, a0.y);
  atomicAdd(ad + 2, a0.z); atomicAdd(ad + 3, a0.w);
  atomicAdd(ad + 4, a1.x); atomicAdd(ad + 5, a1.y);
  atomicAdd(ad + 6, a1.z); atomicAdd(ad + 7, a1.w);
  atomicAdd(as + 0, b0.x); atomicAdd(as + 1, b0.y);
  atomicAdd(as + 2, b0.z); atomicAdd(as + 3, b0.w);
  atomicAdd(as + 4, b1.x); atomicAdd(as + 5, b1.y);
  atomicAdd(as + 6, b1.z); atomicAdd(as + 7, b1.w);
}

// One GIN linear: y = act(in) @ W[32,32] + b, WMMA, optional BN stats gather.
// mode 0: act = in + agg     mode 1: act = relu(BN(in))
__global__ void __launch_bounds__(256)
k_gin_mm(const float* __restrict__ in_a, const float* __restrict__ agg,
         const float* __restrict__ W, const float* __restrict__ bias,
         const float* __restrict__ gamma, const float* __restrict__ beta,
         const float* __restrict__ meanv, const float* __restrict__ invstd,
         float* __restrict__ y_out, float* __restrict__ ssum,
         float* __restrict__ ssq, int mode, int do_stats) {
  // B fragments: Bw[nt][lane][e] = W[16*(lane>>4)+e][16*nt + (lane&15)]
  __shared__ alignas(32) _Float16 Bw[2 * 32 * 16];
  __shared__ float sb[DD], sm[DD], si[DD], sg[DD], sbe[DD];
  int tid = threadIdx.x;
  for (int idx = tid; idx < 1024; idx += 256) {
    int e = idx & 15, ln = (idx >> 4) & 31, nt = idx >> 9;
    int k = 16 * (ln >> 4) + e, n = 16 * nt + (ln & 15);
    Bw[idx] = (_Float16)W[k * DD + n];
  }
  if (tid < DD) {
    sb[tid] = bias[tid];
    if (mode == 1) {
      sm[tid] = meanv[tid]; si[tid] = invstd[tid];
      sg[tid] = gamma[tid]; sbe[tid] = beta[tid];
    }
  }
  __syncthreads();

  int wave = tid >> 5, lane = tid & 31;
  int m = lane & 15, hf = lane >> 4;
  int rowb = blockIdx.x * 128 + wave * 16;
  const float* rp = in_a + (rowb + m) * DD;
  const float* ap = agg + (rowb + m) * DD;

  v16h a;
#pragma unroll
  for (int j = 0; j < 16; ++j) {
    int k = 16 * hf + j;
    float z;
    if (mode == 0) {
      z = rp[k] + ap[k];
    } else {
      float y = rp[k];
      z = fmaxf((y - sm[k]) * si[k] * sg[k] + sbe[k], 0.0f);
    }
    a[j] = (_Float16)z;
  }

#pragma unroll
  for (int nt = 0; nt < 2; ++nt) {
    int n = 16 * nt + (lane & 15);
    float bv = sb[n];
    v8f c;
#pragma unroll
    for (int r = 0; r < 8; ++r) c[r] = bv;
    v16h bvv = *(const v16h*)(Bw + nt * 512 + lane * 16);
    c = wmma_f16(a, bvv, c);
    float s = 0.f, q = 0.f;
#pragma unroll
    for (int r = 0; r < 8; ++r) {
      int mr = rowb + r + 8 * hf;   // D layout: VGPR r -> M = r + 8*hf
      y_out[mr * DD + n] = c[r];
      s += c[r]; q += c[r] * c[r];
    }
    if (do_stats) {
      s += __shfl_xor(s, 16, 32);
      q += __shfl_xor(q, 16, 32);
      if (hf == 0) { atomicAdd(&ssum[n], s); atomicAdd(&ssq[n], q); }
    }
  }
}

__global__ void k_bn_fin(float* __restrict__ st) {  // [sum32|sumsq32|mean32|invstd32]
  int t = threadIdx.x;
  if (t < DD) {
    const float invN = 1.0f / (float)NT;
    float mean = st[t] * invN;
    float var  = st[DD + t] * invN - mean * mean;
    st[2 * DD + t] = mean;
    st[3 * DD + t] = rsqrtf(var + BN_EPS);
  }
}

// Per-graph mean pool: gemb[g][d] = mean_n feats[g*N+n][d]
__global__ void __launch_bounds__(256)
k_pool(const float* __restrict__ feats, float* __restrict__ gemb) {
  int g = blockIdx.x, t = threadIdx.x;
  int d = t & 31, c = t >> 5;
  float s = 0.f;
  for (int n = c; n < NN; n += 8) s += feats[(g * NN + n) * DD + d];
  __shared__ float red[256];
  red[t] = s;
  __syncthreads();
  if (c == 0) {
    float tot = 0.f;
#pragma unroll
    for (int cc = 0; cc < 8; ++cc) tot += red[d + 32 * cc];
    gemb[g * DD + d] = tot * (1.0f / (float)NN);
  }
}

// Critic MLP on 64 graph embeddings (tiny): thread per graph, LDS-staged.
__global__ void __launch_bounds__(64)
k_critic(const float* __restrict__ gemb,
         const float* __restrict__ W1, const float* __restrict__ b1,
         const float* __restrict__ W2, const float* __restrict__ b2,
         const float* __restrict__ W3, const float* __restrict__ b3,
         float* __restrict__ out) {
  __shared__ float sW1[32 * 64];
  __shared__ float sW2[64 * 64];
  __shared__ float sW3[64], sb1[64], sb2[64];
  __shared__ float sH1[64 * 65];
  __shared__ float sH2[64 * 65];
  int t = threadIdx.x;
  for (int i = t; i < 32 * 64; i += 64) sW1[i] = W1[i];
  for (int i = t; i < 64 * 64; i += 64) sW2[i] = W2[i];
  sW3[t] = W3[t]; sb1[t] = b1[t]; sb2[t] = b2[t];
  __syncthreads();
  for (int j = 0; j < 64; ++j) {
    float s = sb1[j];
    for (int k = 0; k < 32; ++k) s += gemb[t * DD + k] * sW1[k * 64 + j];
    sH1[t * 65 + j] = fmaxf(s, 0.f);
  }
  for (int j = 0; j < 64; ++j) {
    float s = sb2[j];
    for (int k = 0; k < 64; ++k) s += sH1[t * 65 + k] * sW2[k * 64 + j];
    sH2[t * 65 + j] = fmaxf(s, 0.f);
  }
  float s = b3[0];
  for (int k = 0; k < 64; ++k) s += sH2[t * 65 + k] * sW3[k];
  out[t] = s;
}

// Actor MLP over all 1M state-action rows: [128 edges/block]x[96->64->64->1].
__global__ void __launch_bounds__(256)
k_actor(const float* __restrict__ gemb, const float* __restrict__ feats,
        const int* __restrict__ src, const int* __restrict__ dst,
        const float* __restrict__ W1, const float* __restrict__ b1,
        const float* __restrict__ W2, const float* __restrict__ b2,
        const float* __restrict__ W3, const float* __restrict__ b3,
        float* __restrict__ logits) {
  __shared__ alignas(32) _Float16 B1[3 * 4 * 32 * 16];  // 12 KB
  __shared__ alignas(32) _Float16 B2[2 * 4 * 32 * 16];  // 8 KB
  __shared__ alignas(32) _Float16 SA[128 * 96];         // 24 KB (reused as H2)
  __shared__ alignas(32) _Float16 H1[128 * 64];         // 16 KB
  __shared__ float sb1[64], sb2[64], sw3[64];
  int tid = threadIdx.x;

  // Stage weights in WMMA B layout: B[ks][nt][lane][e] = W[32*ks+16*(lane>>4)+e][16*nt+(lane&15)]
  for (int idx = tid; idx < 6144; idx += 256) {
    int e = idx & 15, ln = (idx >> 4) & 31, nt = (idx >> 9) & 3, ks = idx >> 11;
    int k = 32 * ks + 16 * (ln >> 4) + e, n = 16 * nt + (ln & 15);
    B1[idx] = (_Float16)W1[k * 64 + n];
  }
  for (int idx = tid; idx < 4096; idx += 256) {
    int e = idx & 15, ln = (idx >> 4) & 31, nt = (idx >> 9) & 3, ks = idx >> 11;
    int k = 32 * ks + 16 * (ln >> 4) + e, n = 16 * nt + (ln & 15);
    B2[idx] = (_Float16)W2[k * 64 + n];
  }
  if (tid < 64) { sb1[tid] = b1[tid]; sb2[tid] = b2[tid]; sw3[tid] = W3[tid]; }

  // Build s_a rows in LDS (f16): [emb[g] | feats[src] | feats[dst]]
  int e0 = blockIdx.x * 128;
  {
    int row = tid >> 1, part = tid & 1;
    int e = e0 + row;
    int g = e >> 14;  // EG = 16384 edges per graph, grouped by graph
    int sN = src[e], dN = dst[e];
    const float* ep = gemb + g * DD;
    const float* sp = feats + sN * DD;
    const float* dp = feats + dN * DD;
    _Float16* sap = SA + row * 96;
    for (int c = part * 48; c < part * 48 + 48; ++c) {
      float v = (c < 32) ? ep[c] : (c < 64) ? sp[c - 32] : dp[c - 64];
      sap[c] = (_Float16)v;
    }
  }
  __syncthreads();

  int wave = tid >> 5, lane = tid & 31;
  int m = lane & 15, hf = lane >> 4;
  int rowb = wave * 16;
  const _Float16* arow = SA + (rowb + m) * 96;

  // Layer 1: [16,96] @ [96,64]
  v8f acc[4];
#pragma unroll
  for (int nt = 0; nt < 4; ++nt) {
    float bv = sb1[16 * nt + (lane & 15)];
#pragma unroll
    for (int r = 0; r < 8; ++r) acc[nt][r] = bv;
  }
#pragma unroll
  for (int ks = 0; ks < 3; ++ks) {
    v16h a = load_a(arow + 32 * ks, hf);
#pragma unroll
    for (int nt = 0; nt < 4; ++nt) {
      v16h bv = *(const v16h*)(B1 + (ks * 4 + nt) * 512 + lane * 16);
      acc[nt] = wmma_f16(a, bv, acc[nt]);
    }
  }
#pragma unroll
  for (int nt = 0; nt < 4; ++nt) {
    int n = 16 * nt + (lane & 15);
#pragma unroll
    for (int r = 0; r < 8; ++r)
      H1[(rowb + r + 8 * hf) * 64 + n] = (_Float16)fmaxf(acc[nt][r], 0.f);
  }
  __syncthreads();

  // Layer 2: [16,64] @ [64,64]
  const _Float16* hrow = H1 + (rowb + m) * 64;
#pragma unroll
  for (int nt = 0; nt < 4; ++nt) {
    float bv = sb2[16 * nt + (lane & 15)];
#pragma unroll
    for (int r = 0; r < 8; ++r) acc[nt][r] = bv;
  }
#pragma unroll
  for (int ks = 0; ks < 2; ++ks) {
    v16h a = load_a(hrow + 32 * ks, hf);
#pragma unroll
    for (int nt = 0; nt < 4; ++nt) {
      v16h bv = *(const v16h*)(B2 + (ks * 4 + nt) * 512 + lane * 16);
      acc[nt] = wmma_f16(a, bv, acc[nt]);
    }
  }
  // ReLU + store h2 into SA region (row stride 96, cols 0..63)
#pragma unroll
  for (int nt = 0; nt < 4; ++nt) {
    int n = 16 * nt + (lane & 15);
#pragma unroll
    for (int r = 0; r < 8; ++r)
      SA[(rowb + r + 8 * hf) * 96 + n] = (_Float16)fmaxf(acc[nt][r], 0.f);
  }
  __syncthreads();

  // Layer 3: scalar head
  if (tid < 128) {
    const _Float16* h2 = SA + tid * 96;
    float s = b3[0];
#pragma unroll
    for (int j = 0; j < 64; ++j) s += (float)h2[j] * sw3[j];
    logits[e0 + tid] = s;
  }
}

// Per-graph softmax over its 16384 edges.
__global__ void __launch_bounds__(256)
k_softmax(const float* __restrict__ logits, float* __restrict__ out) {
  int g = blockIdx.x, t = threadIdx.x;
  const float* lp = logits + g * EG;
  __shared__ float red[256];
  float mx = -1e30f;
  for (int i = t; i < EG; i += 256) mx = fmaxf(mx, lp[i]);
  red[t] = mx;
  __syncthreads();
  for (int s = 128; s > 0; s >>= 1) {
    if (t < s) red[t] = fmaxf(red[t], red[t + s]);
    __syncthreads();
  }
  float gmax = red[0];
  __syncthreads();
  float sum = 0.f;
  for (int i = t; i < EG; i += 256) sum += __expf(lp[i] - gmax);
  red[t] = sum;
  __syncthreads();
  for (int s = 128; s > 0; s >>= 1) {
    if (t < s) red[t] += red[t + s];
    __syncthreads();
  }
  float inv = 1.0f / red[0];
  for (int i = t; i < EG; i += 256) out[g * EG + i] = __expf(lp[i] - gmax) * inv;
}

// ---------------------------------------------------------------------------
extern "C" void kernel_launch(void* const* d_in, const int* in_sizes, int n_in,
                              void* d_out, int out_size, void* d_ws, size_t ws_size,
                              hipStream_t stream) {
  (void)in_sizes; (void)n_in; (void)out_size; (void)ws_size;
  const float* x     = (const float*)d_in[0];
  const int*   ei    = (const int*)d_in[1];
  const float* ginW  = (const float*)d_in[3];   // [3][3][32][32]
  const float* ginB  = (const float*)d_in[4];   // [3][3][32]
  const float* ginG  = (const float*)d_in[5];   // [3][2][32]
  const float* ginBe = (const float*)d_in[6];
  const float* aW1 = (const float*)d_in[7];
  const float* ab1 = (const float*)d_in[8];
  const float* aW2 = (const float*)d_in[9];
  const float* ab2 = (const float*)d_in[10];
  const float* aW3 = (const float*)d_in[11];
  const float* ab3 = (const float*)d_in[12];
  const float* cW1 = (const float*)d_in[13];
  const float* cb1 = (const float*)d_in[14];
  const float* cW2 = (const float*)d_in[15];
  const float* cb2 = (const float*)d_in[16];
  const float* cW3 = (const float*)d_in[17];
  const float* cb3 = (const float*)d_in[18];
  float* out = (float*)d_out;

  // workspace layout (floats)
  float* ws    = (float*)d_ws;
  float* feats = ws;                 // NT*32
  float* agg   = feats + NT * DD;    // NT*32
  float* yA    = agg   + NT * DD;    // NT*32
  float* yB    = yA    + NT * DD;    // NT*32
  float* gemb  = yB    + NT * DD;    // 64*32
  float* stats = gemb  + BG * DD;    // 6 * [sum32|sumsq32|mean32|invstd32]
  float* logits = yA;                // yA dead after GIN; alias
  const int* srcI = ei;
  const int* dstI = ei + ET;

  k_zero<<<dim3(3), dim3(256), 0, stream>>>(stats, 6 * 128);

  for (int l = 0; l < 3; ++l) {
    const float* fin = (l == 0) ? x : feats;
    float* st0 = stats + (l * 2 + 0) * 128;
    float* st1 = stats + (l * 2 + 1) * 128;
    k_zero<<<dim3(NT * DD / 256), dim3(256), 0, stream>>>(agg, NT * DD);
    k_scatter<<<dim3(ET * 4 / 256), dim3(256), 0, stream>>>(fin, srcI, dstI, agg);
    // linear 0: (x+agg)@W0+b0, gather stats for BN0
    k_gin_mm<<<dim3(NT / 128), dim3(256), 0, stream>>>(
        fin, agg, ginW + (l * 3 + 0) * DD * DD, ginB + (l * 3 + 0) * DD,
        nullptr, nullptr, nullptr, nullptr, yA, st0, st0 + 32, 0, 1);
    k_bn_fin<<<dim3(1), dim3(32), 0, stream>>>(st0);
    // linear 1: relu(BN0(y0))@W1+b1, gather stats for BN1
    k_gin_mm<<<dim3(NT / 128), dim3(256), 0, stream>>>(
        yA, nullptr, ginW + (l * 3 + 1) * DD * DD, ginB + (l * 3 + 1) * DD,
        ginG + (l * 2 + 0) * DD, ginBe + (l * 2 + 0) * DD,
        st0 + 64, st0 + 96, yB, st1, st1 + 32, 1, 1);
    k_bn_fin<<<dim3(1), dim3(32), 0, stream>>>(st1);
    // linear 2: relu(BN1(y1))@W2+b2 -> feats
    k_gin_mm<<<dim3(NT / 128), dim3(256), 0, stream>>>(
        yB, nullptr, ginW + (l * 3 + 2) * DD * DD, ginB + (l * 3 + 2) * DD,
        ginG + (l * 2 + 1) * DD, ginBe + (l * 2 + 1) * DD,
        st1 + 64, st1 + 96, feats, nullptr, nullptr, 1, 0);
  }

  k_pool<<<dim3(BG), dim3(256), 0, stream>>>(feats, gemb);
  k_critic<<<dim3(1), dim3(64), 0, stream>>>(gemb, cW1, cb1, cW2, cb2, cW3, cb3,
                                             out + ET);
  k_actor<<<dim3(ET / 128), dim3(256), 0, stream>>>(
      gemb, feats, srcI, dstI, aW1, ab1, aW2, ab2, aW3, ab3, logits);
  k_softmax<<<dim3(BG), dim3(256), 0, stream>>>(logits, out);
}
